// ConvTransformerEncoderLayer_35416300323415
// MI455X (gfx1250) — matrix operand order repaired and checked by reading, not compiled, not run on hardware
//
#include <hip/hip_runtime.h>
#include <hip/hip_bf16.h>

typedef __attribute__((ext_vector_type(16))) _Float16 v16h;
typedef __attribute__((ext_vector_type(8)))  float    v8f;

#define LPAD 40   // LDS row stride in halfs (80B) -> conflict-free 16-row column walks

// ---------------------------------------------------------------------------
// Weight repack: W[cout][cin][3][3] f32  ->  Wt[cout][tap*Cin + cin] f16
// ---------------------------------------------------------------------------
__global__ __launch_bounds__(256) void repack_weights(
    const float* __restrict__ w, _Float16* __restrict__ wt, int Cout, int Cin)
{
    int i = blockIdx.x * 256 + threadIdx.x;
    int K = 9 * Cin;
    if (i >= Cout * K) return;
    int cout = i / K;
    int k    = i - cout * K;
    int tap  = k / Cin;
    int cin  = k - tap * Cin;
    wt[i] = (_Float16)w[((size_t)cout * Cin + cin) * 9 + tap];
}

// ---------------------------------------------------------------------------
// 3x3 SAME conv as implicit-im2col WMMA GEMM, double-buffered K pipeline.
//   x  : (B, Cin, 8, 8) f32        y : (B, Cout, 8, 8) f32
//   wt : (Cout, 9*Cin)  f16 packed
// Grid: (B, Cout/64), 256 threads = 8 waves.
// WG computes 64 pixels x 64 couts; wave (wm,wn) computes 16x32 via 2 accums.
// A: branchless clamped-halo loads prefetched into regs one chunk ahead.
// B: CDNA5 async global->LDS DMA (ASYNCcnt), one chunk ahead.
// One barrier per 32-K chunk.
// ---------------------------------------------------------------------------
__global__ __launch_bounds__(256) void conv3x3_wmma(
    const float* __restrict__ x, const _Float16* __restrict__ wt,
    const float* __restrict__ bias, float* __restrict__ y,
    int Cin, int Cout, int relu)
{
    __shared__ _Float16 Alds[2][64 * LPAD];  // [buf][pixel][k]
    __shared__ _Float16 Blds[2][64 * LPAD];  // [buf][cout][k]

    const int tid  = threadIdx.x;
    const int lane = tid & 31;
    const int wv   = tid >> 5;       // 0..7
    const int wm   = wv & 3;         // 4 M-subtiles of 16 pixels
    const int wn   = wv >> 2;        // 2 N-subtiles of 32 couts
    const int b    = blockIdx.x;     // one image == one 64-pixel M tile
    const int n0   = blockIdx.y * 64;
    const int Ktot = 9 * Cin;

    v8f acc0 = {}; v8f acc1 = {};

    // staging assignments (256 threads, 2048 halfs each for A and B per chunk)
    const int ap = tid & 63;              // pixel 0..63
    const int ph = ap >> 3, pw_ = ap & 7;
    const int ak = (tid >> 6) * 8;        // k sub-block {0,8,16,24}
    const int bn = tid >> 2;              // cout row 0..63
    const int bk = (tid & 3) * 8;         // k sub-block {0,8,16,24}

    // fragment addressing (contiguous k = 16*half + s, same order for A and B)
    const int hsel  = lane >> 4;          // lane half select
    const int arow  = wm * 16 + (lane & 15);
    const int brow0 = wn * 32 + (lane & 15);
    const int bko   = hsel * 16;

    // LDS byte offsets of this thread's B staging slot (low 32 bits = LDS offset)
    const unsigned bOff0 = (unsigned)(size_t)&Blds[0][bn * LPAD + bk];
    const unsigned bOff1 = (unsigned)(size_t)&Blds[1][bn * LPAD + bk];

    const float*    xb  = x  + (size_t)b * Cin * 64;
    const _Float16* wtb = wt + (size_t)(n0 + bn) * Ktot + bk;

    const int nchunks = 9 * (Cin >> 5);

    // ---- prologue: prefetch chunk 0 (tap 0, cc 0) ----
    float areg[8];
    {
        const int hy = ph - 1, wx = pw_ - 1;
        const bool inb = (hy >= 0) & (wx >= 0);          // tap 0: dh=dw=-1
        const int hyc = hy < 0 ? 0 : hy, wxc = wx < 0 ? 0 : wx;
        const float* sA = xb + hyc * 8 + wxc;
        #pragma unroll
        for (int j = 0; j < 8; ++j) {
            float v = sA[(size_t)(ak + j) * 64];
            areg[j] = inb ? v : 0.0f;
        }
        asm volatile("global_load_async_to_lds_b128 %0, %1, off"
                     :: "v"(bOff0), "v"(wtb) : "memory");
    }

    int tapn = 0, ccn = 32;                // params of chunk 1
    if (ccn == Cin) { ccn = 0; tapn = 1; }
    int p = 0;

    for (int c = 0; c < nchunks; ++c) {
        // ---- commit A(c) to LDS buf p ----
        union { _Float16 h[8]; uint4 u; } at;
        #pragma unroll
        for (int j = 0; j < 8; ++j) at.h[j] = (_Float16)areg[j];
        *(uint4*)&Alds[p][ap * LPAD + ak] = at.u;

        asm volatile("s_wait_asynccnt 0x0" ::: "memory");   // B(c) resident
        __syncthreads();                                    // buf p fully staged

        // ---- prefetch chunk c+1 into buf p^1 (overlaps with WMMAs below) ----
        if (c + 1 < nchunks) {
            const int dh = tapn / 3 - 1, dw = tapn % 3 - 1;
            const int hy = ph + dh, wx = pw_ + dw;
            const bool inb = (hy >= 0) & (hy < 8) & (wx >= 0) & (wx < 8);
            const int hyc = hy < 0 ? 0 : (hy > 7 ? 7 : hy);
            const int wxc = wx < 0 ? 0 : (wx > 7 ? 7 : wx);
            const float* sA = xb + hyc * 8 + wxc;
            #pragma unroll
            for (int j = 0; j < 8; ++j) {
                float v = sA[(size_t)(tapn * 0 + ccn + ak + j) * 64];
                areg[j] = inb ? v : 0.0f;
            }
            asm volatile("global_load_async_to_lds_b128 %0, %1, off"
                         :: "v"(p ? bOff0 : bOff1),
                            "v"(wtb + (size_t)tapn * Cin + ccn) : "memory");
            ccn += 32;
            if (ccn == Cin) { ccn = 0; ++tapn; }
        }

        // ---- fragments from buf p + WMMA ----
        union { uint4 u[2]; v16h v; } af, bf0, bf1;
        af.u[0]  = *(const uint4*)&Alds[p][arow * LPAD + hsel * 16];
        af.u[1]  = *(const uint4*)&Alds[p][arow * LPAD + hsel * 16 + 8];
        bf0.u[0] = *(const uint4*)&Blds[p][brow0 * LPAD + bko];
        bf0.u[1] = *(const uint4*)&Blds[p][brow0 * LPAD + bko + 8];
        bf1.u[0] = *(const uint4*)&Blds[p][(brow0 + 16) * LPAD + bko];
        bf1.u[1] = *(const uint4*)&Blds[p][(brow0 + 16) * LPAD + bko + 8];

        acc0 = __builtin_amdgcn_wmma_f32_16x16x32_f16(
                   false, af.v, false, bf0.v, (short)0, acc0, false, false);
        acc1 = __builtin_amdgcn_wmma_f32_16x16x32_f16(
                   false, af.v, false, bf1.v, (short)0, acc1, false, false);
        p ^= 1;
    }

    // ---- epilogue: C layout -> NCHW, + bias, optional ReLU ----
    const int mbase = wm * 16 + hsel * 8;
    const int nc0   = n0 + wn * 32 + (lane & 15);
    const float bv0 = bias[nc0];
    const float bv1 = bias[nc0 + 16];
    float* yb = y + (size_t)b * Cout * 64;
    #pragma unroll
    for (int r = 0; r < 8; ++r) {
        int m = mbase + r;
        float v0 = acc0[r] + bv0;
        float v1 = acc1[r] + bv1;
        if (relu) { v0 = fmaxf(v0, 0.0f); v1 = fmaxf(v1, 0.0f); }
        yb[(size_t)nc0 * 64 + m]        = v0;
        yb[(size_t)(nc0 + 16) * 64 + m] = v1;
    }
}

// ---------------------------------------------------------------------------
// Attention over 32 memory slots. One WG per (b, head). Pure streaming.
//   kqv : (B,1536,8,8)  ck/cv : (B,32,8,4096)  pos_w : (32,32768)  pos_b:(32,8)
//   ret_k/ret_v : (B,32,8,4096)   attn_out : (B,512,8,8)
// ---------------------------------------------------------------------------
__global__ __launch_bounds__(256) void attention_kernel(
    const float* __restrict__ kqv, const float* __restrict__ ck,
    const float* __restrict__ cv, const unsigned char* __restrict__ mask,
    const float* __restrict__ pos_w, const float* __restrict__ pos_b,
    float* __restrict__ ret_k, float* __restrict__ ret_v,
    float* __restrict__ attn_out)
{
    const int bh = blockIdx.x;               // 0..511
    const int b = bh >> 3, nh = bh & 7;
    const int tid = threadIdx.x;
    const int lane = tid & 31, wv = tid >> 5;

    __shared__ float red[8 * 32];
    __shared__ float slog[32];

    float part[32];
    #pragma unroll
    for (int m = 0; m < 32; ++m) part[m] = 0.0f;

    const float* kqvb = kqv + (size_t)b * 1536 * 64;
    const size_t mem_base = (((size_t)b * 32 + 1) * 8 + nh) * 4096;  // m=0 slot
    const size_t ret_base = (((size_t)b * 32)     * 8 + nh) * 4096;

    for (int d = tid; d < 4096; d += 256) {
        const int ch = d >> 6, pix = d & 63;
        const float qv = kqvb[(512  + nh * 64 + ch) * 64 + pix] * 0.015625f; // 1/sqrt(4096)
        const float kc = kqvb[(       nh * 64 + ch) * 64 + pix];
        const float vc = kqvb[(1024 + nh * 64 + ch) * 64 + pix];
        const float* pwd = pos_w + (size_t)nh * 4096 + d;   // + m*32768
        const float* ckd = ck + mem_base + d;               // + m*32768
        const float* cvd = cv + mem_base + d;
        float* rkd = ret_k + ret_base + d;
        float* rvd = ret_v + ret_base + d;
        #pragma unroll 4
        for (int m = 0; m < 32; ++m) {
            float km = (m < 31) ? ckd[(size_t)m * 32768] : kc;
            float vm = (m < 31) ? cvd[(size_t)m * 32768] : vc;
            km += pwd[(size_t)m * 32768];
            rkd[(size_t)m * 32768] = km;
            rvd[(size_t)m * 32768] = vm;
            part[m] = fmaf(qv, km, part[m]);
        }
    }

    // wave reduce then cross-wave reduce
    #pragma unroll
    for (int m = 0; m < 32; ++m) {
        float v = part[m];
        for (int off = 16; off > 0; off >>= 1) v += __shfl_xor(v, off, 32);
        if (lane == 0) red[wv * 32 + m] = v;
    }
    __syncthreads();
    if (tid < 32) {
        float s = 0.0f;
        #pragma unroll
        for (int w8 = 0; w8 < 8; ++w8) s += red[w8 * 32 + tid];
        float mv = (tid == 31) ? 5.0f
                               : (mask[bh * 32 + tid] ? -__builtin_inff() : 0.0f);
        slog[tid] = s + mv + pos_b[tid * 8 + nh];
    }
    __syncthreads();

    // redundant per-thread softmax over 32 logits (cheap)
    float mx = -__builtin_inff();
    #pragma unroll
    for (int m = 0; m < 32; ++m) mx = fmaxf(mx, slog[m]);
    float wgt[32]; float ssum = 0.0f;
    #pragma unroll
    for (int m = 0; m < 32; ++m) { wgt[m] = __expf(slog[m] - mx); ssum += wgt[m]; }
    const float inv = 1.0f / ssum;
    #pragma unroll
    for (int m = 0; m < 32; ++m) wgt[m] *= inv;

    // attn = sum_m w[m] * v[m]; re-read ret_v (same-thread store->load, ordered)
    for (int d = tid; d < 4096; d += 256) {
        const float* rvd = ret_v + ret_base + d;
        float a = 0.0f;
        #pragma unroll 4
        for (int m = 0; m < 32; ++m) a = fmaf(wgt[m], rvd[(size_t)m * 32768], a);
        attn_out[(size_t)b * 32768 + (size_t)nh * 4096 + d] = a;
    }
}

// ---------------------------------------------------------------------------
// out = layernorm(a + r) over 512*64 elems per sample, elementwise affine.
// ---------------------------------------------------------------------------
__global__ __launch_bounds__(256) void residual_layernorm(
    const float* __restrict__ a, const float* __restrict__ r,
    const float* __restrict__ w, const float* __restrict__ bb,
    float* __restrict__ out)
{
    const int b = blockIdx.x;
    const int tid = threadIdx.x;
    const int lane = tid & 31, wv = tid >> 5;
    __shared__ float rs[8], rs2[8];

    const float* ab = a + (size_t)b * 32768;
    const float* rb = r + (size_t)b * 32768;
    float s = 0.0f, s2 = 0.0f;
    for (int i = tid; i < 32768; i += 256) {
        float v = ab[i] + rb[i];
        s += v; s2 += v * v;
    }
    for (int off = 16; off > 0; off >>= 1) {
        s  += __shfl_xor(s,  off, 32);
        s2 += __shfl_xor(s2, off, 32);
    }
    if (lane == 0) { rs[wv] = s; rs2[wv] = s2; }
    __syncthreads();
    float ts = 0.0f, ts2 = 0.0f;
    #pragma unroll
    for (int w8 = 0; w8 < 8; ++w8) { ts += rs[w8]; ts2 += rs2[w8]; }
    const float mean = ts * (1.0f / 32768.0f);
    const float var  = ts2 * (1.0f / 32768.0f) - mean * mean;
    const float rstd = rsqrtf(var + 1e-5f);

    float* ob = out + (size_t)b * 32768;
    for (int i = tid; i < 32768; i += 256) {
        float v = ab[i] + rb[i];
        ob[i] = (v - mean) * rstd * w[i] + bb[i];
    }
}

// ---------------------------------------------------------------------------
extern "C" void kernel_launch(void* const* d_in, const int* in_sizes, int n_in,
                              void* d_out, int out_size, void* d_ws, size_t ws_size,
                              hipStream_t stream) {
    const float* x      = (const float*)d_in[0];
    const unsigned char* mask = (const unsigned char*)d_in[1];
    const float* ck     = (const float*)d_in[2];
    const float* cv     = (const float*)d_in[3];
    const float* proj_w = (const float*)d_in[4];
    const float* proj_b = (const float*)d_in[5];
    const float* out_w  = (const float*)d_in[6];
    const float* out_b  = (const float*)d_in[7];
    const float* lin1_w = (const float*)d_in[8];
    const float* lin1_b = (const float*)d_in[9];
    const float* lin2_w = (const float*)d_in[10];
    const float* lin2_b = (const float*)d_in[11];
    const float* n1_w   = (const float*)d_in[12];
    const float* n1_b   = (const float*)d_in[13];
    const float* n2_w   = (const float*)d_in[14];
    const float* n2_b   = (const float*)d_in[15];
    const float* pos_w  = (const float*)d_in[16];
    const float* pos_b  = (const float*)d_in[17];

    float* out   = (float*)d_out;                 // (1,64,512,8,8)
    float* ret_k = out + 2097152;                 // (64,32,8,4096)
    float* ret_v = ret_k + 67108864;

    char* p = (char*)d_ws;
    auto alloc = [&](size_t bytes) -> char* {
        char* q = p; p += (bytes + 255) & ~(size_t)255; return q;
    };
    _Float16* wt_proj = (_Float16*)alloc((size_t)1536 * 4608  * 2);
    _Float16* wt_out  = (_Float16*)alloc((size_t)512  * 4608  * 2);
    _Float16* wt_lin1 = (_Float16*)alloc((size_t)2048 * 4608  * 2);
    _Float16* wt_lin2 = (_Float16*)alloc((size_t)512  * 18432 * 2);
    float* kqv   = (float*)alloc((size_t)64 * 1536 * 64 * 4);
    float* attn  = (float*)alloc((size_t)64 * 512  * 64 * 4);
    float* conv1 = (float*)alloc((size_t)64 * 512  * 64 * 4);
    float* out1  = (float*)alloc((size_t)64 * 512  * 64 * 4);
    float* ff1   = (float*)alloc((size_t)64 * 2048 * 64 * 4);
    float* ff2   = (float*)alloc((size_t)64 * 512  * 64 * 4);

    // 1. pack conv weights to f16 [cout][tap*Cin+cin]
    repack_weights<<<(1536 * 4608  + 255) / 256, 256, 0, stream>>>(proj_w, wt_proj, 1536, 512);
    repack_weights<<<(512  * 4608  + 255) / 256, 256, 0, stream>>>(out_w,  wt_out,  512,  512);
    repack_weights<<<(2048 * 4608  + 255) / 256, 256, 0, stream>>>(lin1_w, wt_lin1, 2048, 512);
    repack_weights<<<(512  * 18432 + 255) / 256, 256, 0, stream>>>(lin2_w, wt_lin2, 512,  2048);

    // 2. proj conv -> kqv
    conv3x3_wmma<<<dim3(64, 1536 / 64), 256, 0, stream>>>(x, wt_proj, proj_b, kqv, 512, 1536, 0);
    // 3. attention (writes ret_k/ret_v into d_out, attn into ws)
    attention_kernel<<<512, 256, 0, stream>>>(kqv, ck, cv, mask, pos_w, pos_b, ret_k, ret_v, attn);
    // 4. out conv, residual + LN1
    conv3x3_wmma<<<dim3(64, 512 / 64), 256, 0, stream>>>(attn, wt_out, out_b, conv1, 512, 512, 0);
    residual_layernorm<<<64, 256, 0, stream>>>(conv1, x, n1_w, n1_b, out1);
    // 5. FF: lin1 (+ReLU), lin2, residual + LN2 -> final out
    conv3x3_wmma<<<dim3(64, 2048 / 64), 256, 0, stream>>>(out1, wt_lin1, lin1_b, ff1, 512, 2048, 1);
    conv3x3_wmma<<<dim3(64, 512 / 64), 256, 0, stream>>>(ff1, wt_lin2, lin2_b, ff2, 2048, 512, 0);
    residual_layernorm<<<64, 256, 0, stream>>>(ff2, out1, n2_w, n2_b, out);
}
